// GatingMixedDecoder_54331336294541
// MI455X (gfx1250) — matrix operand order repaired and checked by reading, not compiled, not run on hardware
//
#include <hip/hip_runtime.h>
#include <hip/hip_bf16.h>
#include <stdint.h>

// MI455X / gfx1250: wave32, v_wmma_f32_16x16x32_bf16 (f32 accumulate).
// Mixture-of-experts decoder, compute-bound (~2000 FLOP/byte): weights are
// pre-transposed to bf16 [E,N,K] once per layer; GEMM hot loop double-buffers
// LDS tiles via GLOBAL_LOAD_ASYNC_TO_LDS_B128 with counted S_WAIT_ASYNCCNT so
// the async DMA for stage i+1 overlaps the WMMAs of stage i.
// Tiles: block 64x128, BK=64, wave 32x32 (8 wmma/stage).

typedef __attribute__((ext_vector_type(4)))  __bf16 v4bf;
typedef __attribute__((ext_vector_type(8)))  __bf16 v8bf;
typedef __attribute__((ext_vector_type(16))) __bf16 v16bf;
typedef __attribute__((ext_vector_type(8)))  float  v8f;

#define NEG_SLOPE 0.01f
#define LN_EPS    1e-5f
#define LDS_PAD   72   // 64 + 8 bf16: 144B rows -> 16B aligned, bank-conflict-free

__device__ __forceinline__ v16bf cat8(v8bf lo, v8bf hi) {
  return __builtin_shufflevector(lo, hi, 0,1,2,3,4,5,6,7,8,9,10,11,12,13,14,15);
}
__device__ __forceinline__ float leakyf(float v) { return v > 0.f ? v : NEG_SLOPE * v; }

// Block-wide sum of two values (256 threads = 8 wave32).
__device__ __forceinline__ void block_reduce2(float& a, float& b) {
  #pragma unroll
  for (int m = 16; m; m >>= 1) { a += __shfl_xor(a, m, 32); b += __shfl_xor(b, m, 32); }
  __shared__ float sa[8], sb[8];
  int lane = threadIdx.x & 31, wv = threadIdx.x >> 5;
  if (lane == 0) { sa[wv] = a; sb[wv] = b; }
  __syncthreads();
  float ta = 0.f, tb = 0.f;
  #pragma unroll
  for (int i = 0; i < 8; ++i) { ta += sa[i]; tb += sb[i]; }
  a = ta; b = tb;
}

// ---------------------------------------------------------------------------
// transpose_bf16: Wout[e][n][k] = (bf16) Win[e][k][n].  32x32 LDS tiles.
// grid: (K/32, N/32, E), 256 threads.
// ---------------------------------------------------------------------------
__global__ __launch_bounds__(256) void transpose_bf16(
    const float* __restrict__ Win, __bf16* __restrict__ Wout, int K, int N) {
  __shared__ float t[32][33];
  const int tid = threadIdx.x;
  const int k0 = blockIdx.x * 32, n0 = blockIdx.y * 32;
  const size_t eoff = (size_t)blockIdx.z * (size_t)K * (size_t)N;
  const float* We = Win + eoff;
  __bf16* Oe = Wout + eoff;

  int r = tid >> 3, cg = (tid & 7) * 4;
  float4 v = *(const float4*)&We[(size_t)(k0 + r) * N + n0 + cg];
  t[r][cg + 0] = v.x; t[r][cg + 1] = v.y; t[r][cg + 2] = v.z; t[r][cg + 3] = v.w;
  __syncthreads();
  v4bf o = { (__bf16)t[cg + 0][r], (__bf16)t[cg + 1][r],
             (__bf16)t[cg + 2][r], (__bf16)t[cg + 3][r] };
  *(v4bf*)&Oe[(size_t)(n0 + r) * K + k0 + cg] = o;
}

// ---------------------------------------------------------------------------
// prep: zc = concat(z,c) -> bf16   and   x0 = LayerNorm(zc) -> bf16
// ---------------------------------------------------------------------------
__global__ __launch_bounds__(256) void prep_kernel(
    const float* __restrict__ z, const float* __restrict__ c,
    const float* __restrict__ g, const float* __restrict__ bta,
    __bf16* __restrict__ zc, __bf16* __restrict__ x0) {
  const int row = blockIdx.x, tid = threadIdx.x;
  float v[5]; float s = 0.f, s2 = 0.f;
  #pragma unroll
  for (int i = 0; i < 5; ++i) {
    int j = tid + i * 256;
    float val = (j < 768) ? z[(size_t)row * 768 + j] : c[(size_t)row * 512 + (j - 768)];
    v[i] = val; s += val; s2 += val * val;
    zc[(size_t)row * 1280 + j] = (__bf16)val;
  }
  block_reduce2(s, s2);
  float mu = s * (1.0f / 1280.0f);
  float var = s2 * (1.0f / 1280.0f) - mu * mu;
  float rs = rsqrtf(var + LN_EPS);
  #pragma unroll
  for (int i = 0; i < 5; ++i) {
    int j = tid + i * 256;
    x0[(size_t)row * 1280 + j] = (__bf16)((v[i] - mu) * rs * g[j] + bta[j]);
  }
}

// ---------------------------------------------------------------------------
// ln_concat: x = LayerNorm(concat(z, layer_out)) -> bf16 [B,1792]
// ---------------------------------------------------------------------------
__global__ __launch_bounds__(256) void ln_concat_kernel(
    const float* __restrict__ z, const float* __restrict__ lo,
    const float* __restrict__ g, const float* __restrict__ bta,
    __bf16* __restrict__ xout) {
  const int row = blockIdx.x, tid = threadIdx.x;
  float v[7]; float s = 0.f, s2 = 0.f;
  #pragma unroll
  for (int i = 0; i < 7; ++i) {
    int j = tid + i * 256;
    float val = (j < 768) ? z[(size_t)row * 768 + j] : lo[(size_t)row * 1024 + (j - 768)];
    v[i] = val; s += val; s2 += val * val;
  }
  block_reduce2(s, s2);
  float mu = s * (1.0f / 1792.0f);
  float var = s2 * (1.0f / 1792.0f) - mu * mu;
  float rs = rsqrtf(var + LN_EPS);
  #pragma unroll
  for (int i = 0; i < 7; ++i) {
    int j = tid + i * 256;
    xout[(size_t)row * 1792 + j] = (__bf16)((v[i] - mu) * rs * g[j] + bta[j]);
  }
}

// ---------------------------------------------------------------------------
// mixgemm: out[b,n] = post( [resid +] sum_e coeff[b,e] * (X[b,:] @ W_e + b_e[n]) )
// X bf16 [B,K]; WT bf16 [E,N,K] (pre-transposed).  Block 64(M) x 128(N), BK=64.
// 8 waves as 2(m) x 4(n); wave tile 32x32 -> 4 wmma per 32-K step, 8 per stage.
// Double-buffered async LDS staging: stage i+1 DMA overlaps stage i WMMA.
// NE=1 -> plain GEMM (gate layers).  Requires K % 128 == 0.
// ---------------------------------------------------------------------------
template<int NE, bool RES, bool LEAKY, bool OUTB>
__global__ __launch_bounds__(256) void mixgemm(
    const __bf16* __restrict__ X, const __bf16* __restrict__ WT,
    const float* __restrict__ Bv, const float* __restrict__ coeff,
    const float* __restrict__ resid, float* __restrict__ outF,
    __bf16* __restrict__ outB, int K, int N) {
  __shared__ __align__(16) __bf16 xt[2][64][LDS_PAD];    // X tiles [buf][m][k]
  __shared__ __align__(16) __bf16 wt[2][128][LDS_PAD];   // W tiles [buf][n][k]
  __shared__ float ct[64][NE];                           // coeff tile

  const int tid  = threadIdx.x;
  const int lane = tid & 31, wv = tid >> 5;
  const int wm = wv >> 2, wn = wv & 3;                // 2 x 4 wave grid
  const int m0 = blockIdx.x * 64, n0 = blockIdx.y * 128;
  const int l15 = lane & 15, lh = lane >> 4;

  if (NE > 1) {
    for (int i = tid; i < 64 * NE; i += 256)
      ct[i / NE][i % NE] = coeff[(size_t)(m0 + i / NE) * NE + (i % NE)];
  }

  // staging thread mapping (stage-invariant)
  const int xr = tid >> 2, xc = (tid & 3) * 16;       // X: 64 rows x 64 k
  const int wr = tid >> 1, wc = (tid & 1) * 32;       // W: 128 rows x 64 k
  // workgroup-relative LDS byte offsets (low 32 bits of the generic address)
  const unsigned xl0 = (unsigned)(uintptr_t)&xt[0][xr][xc];
  const unsigned xl1 = (unsigned)(uintptr_t)&xt[1][xr][xc];
  const unsigned wl0 = (unsigned)(uintptr_t)&wt[0][wr][wc];
  const unsigned wl1 = (unsigned)(uintptr_t)&wt[1][wr][wc];

  // CDNA5 async DMA: memory -> LDS directly, tracked by ASYNCcnt.
  // IOFFSET applies to both LDS and global addresses (same chunk layout).
  auto issue = [&](unsigned xl, unsigned wl, const __bf16* xp, const __bf16* wp) {
    asm volatile(
      "global_load_async_to_lds_b128 %0, %2, off\n\t"
      "global_load_async_to_lds_b128 %0, %2, off offset:16\n\t"
      "global_load_async_to_lds_b128 %1, %3, off\n\t"
      "global_load_async_to_lds_b128 %1, %3, off offset:16\n\t"
      "global_load_async_to_lds_b128 %1, %3, off offset:32\n\t"
      "global_load_async_to_lds_b128 %1, %3, off offset:48"
      :: "v"(xl), "v"(wl),
         "v"((unsigned long long)(uintptr_t)xp),
         "v"((unsigned long long)(uintptr_t)wp)
      : "memory");
  };

  v8f f00 = {}, f01 = {}, f10 = {}, f11 = {};
  const int ncol0 = n0 + wn * 32 + l15;
  const int ncol1 = ncol0 + 16;

  for (int e = 0; e < NE; ++e) {
    v8f p00 = {}, p01 = {}, p10 = {}, p11 = {};
    const __bf16* xg = X + (size_t)(m0 + xr) * K + xc;
    const __bf16* wg = WT + ((size_t)e * N + n0 + wr) * K + wc;

    auto compute = [&](int buf) {
      #pragma unroll
      for (int kk = 0; kk < 64; kk += 32) {
        // A frags (16x32): lane<16 K{kk+0..7, kk+16..23}; lane>=16 {+8, +24}
        int ra = wm * 32 + l15;
        int c0 = kk + lh * 8, c1 = kk + 16 + lh * 8;
        v16bf a0 = cat8(*(const v8bf*)&xt[buf][ra][c0],      *(const v8bf*)&xt[buf][ra][c1]);
        v16bf a1 = cat8(*(const v8bf*)&xt[buf][ra + 16][c0], *(const v8bf*)&xt[buf][ra + 16][c1]);
        // B frags (32x16): lane<16 K{kk+0..7, kk+16..23} at n=lane; lane>=16 {+8, +24}
        int nb = wn * 32 + l15, kb = kk + lh * 8;
        v16bf b0 = cat8(*(const v8bf*)&wt[buf][nb][kb],      *(const v8bf*)&wt[buf][nb][kb + 16]);
        v16bf b1 = cat8(*(const v8bf*)&wt[buf][nb + 16][kb], *(const v8bf*)&wt[buf][nb + 16][kb + 16]);

        p00 = __builtin_amdgcn_wmma_f32_16x16x32_bf16(false, a0, false, b0, (short)0, p00, false, false);
        p01 = __builtin_amdgcn_wmma_f32_16x16x32_bf16(false, a0, false, b1, (short)0, p01, false, false);
        p10 = __builtin_amdgcn_wmma_f32_16x16x32_bf16(false, a1, false, b0, (short)0, p10, false, false);
        p11 = __builtin_amdgcn_wmma_f32_16x16x32_bf16(false, a1, false, b1, (short)0, p11, false, false);
      }
    };

    // prologue: stage 0 -> buf0
    issue(xl0, wl0, xg, wg); xg += 64; wg += 64;

    for (int ks = 0; ks < K; ks += 128) {
      // ---- stage A (buf0): issue next tile into buf1, wait for buf0 tile ----
      issue(xl1, wl1, xg, wg); xg += 64; wg += 64;
      asm volatile("s_wait_asynccnt 0x6" ::: "memory");   // 6 newest still in flight
      __syncthreads();                                    // buf0 published
      compute(0);
      __syncthreads();                                    // buf0 free
      // ---- stage B (buf1) ----
      if (ks + 128 < K) {
        issue(xl0, wl0, xg, wg); xg += 64; wg += 64;
        __builtin_prefetch((const void*)wg, 0, 1);        // warm L2 one stage ahead
        __builtin_prefetch((const void*)xg, 0, 1);
        asm volatile("s_wait_asynccnt 0x6" ::: "memory");
      } else {
        asm volatile("s_wait_asynccnt 0x0" ::: "memory"); // drain last tile
      }
      __syncthreads();                                    // buf1 published
      compute(1);
      __syncthreads();                                    // buf1 free
    }

    // fold expert e: fin += coeff[row,e] * (partial + bias_e[n])
    float bv0 = Bv[(size_t)e * N + ncol0];
    float bv1 = Bv[(size_t)e * N + ncol1];
    #pragma unroll
    for (int r = 0; r < 8; ++r) {
      int rl0 = wm * 32 + r + 8 * lh;       // C layout: row = r + 8*(lane>=16)
      int rl1 = rl0 + 16;
      float cf0 = (NE > 1) ? ct[rl0][e] : 1.0f;
      float cf1 = (NE > 1) ? ct[rl1][e] : 1.0f;
      f00[r] += cf0 * (p00[r] + bv0);
      f01[r] += cf0 * (p01[r] + bv1);
      f10[r] += cf1 * (p10[r] + bv0);
      f11[r] += cf1 * (p11[r] + bv1);
    }
  }

  #pragma unroll
  for (int r = 0; r < 8; ++r) {
    int row0 = m0 + wm * 32 + r + 8 * lh;
    int row1 = row0 + 16;
    float v00 = f00[r], v01 = f01[r], v10 = f10[r], v11 = f11[r];
    if (RES) {
      v00 += resid[(size_t)row0 * N + ncol0];
      v01 += resid[(size_t)row0 * N + ncol1];
      v10 += resid[(size_t)row1 * N + ncol0];
      v11 += resid[(size_t)row1 * N + ncol1];
    }
    if (LEAKY) { v00 = leakyf(v00); v01 = leakyf(v01); v10 = leakyf(v10); v11 = leakyf(v11); }
    if (OUTB) {
      outB[(size_t)row0 * N + ncol0] = (__bf16)v00;
      outB[(size_t)row0 * N + ncol1] = (__bf16)v01;
      outB[(size_t)row1 * N + ncol0] = (__bf16)v10;
      outB[(size_t)row1 * N + ncol1] = (__bf16)v11;
    } else {
      outF[(size_t)row0 * N + ncol0] = v00;
      outF[(size_t)row0 * N + ncol1] = v01;
      outF[(size_t)row1 * N + ncol0] = v10;
      outF[(size_t)row1 * N + ncol1] = v11;
    }
  }
}

// ---------------------------------------------------------------------------
// gate_softmax: coeff = softmax(g2 @ gW2 + gb2), one wave per row.
// ---------------------------------------------------------------------------
__global__ __launch_bounds__(256) void gate_softmax(
    const __bf16* __restrict__ g2, const float* __restrict__ gW2,
    const float* __restrict__ gb2, float* __restrict__ coeff) {
  const int row = blockIdx.x * 8 + (threadIdx.x >> 5);
  const int lane = threadIdx.x & 31;
  float s0=0,s1=0,s2=0,s3=0,s4=0,s5=0,s6=0,s7=0;
  for (int k = lane; k < 512; k += 32) {
    float xv = (float)g2[(size_t)row * 512 + k];
    const float* wp = gW2 + (size_t)k * 8;
    float4 wa = *(const float4*)wp;
    float4 wb = *(const float4*)(wp + 4);
    s0 += xv * wa.x; s1 += xv * wa.y; s2 += xv * wa.z; s3 += xv * wa.w;
    s4 += xv * wb.x; s5 += xv * wb.y; s6 += xv * wb.z; s7 += xv * wb.w;
  }
  #pragma unroll
  for (int m = 16; m; m >>= 1) {
    s0 += __shfl_xor(s0, m, 32); s1 += __shfl_xor(s1, m, 32);
    s2 += __shfl_xor(s2, m, 32); s3 += __shfl_xor(s3, m, 32);
    s4 += __shfl_xor(s4, m, 32); s5 += __shfl_xor(s5, m, 32);
    s6 += __shfl_xor(s6, m, 32); s7 += __shfl_xor(s7, m, 32);
  }
  if (lane == 0) {
    float sv[8] = { s0 + gb2[0], s1 + gb2[1], s2 + gb2[2], s3 + gb2[3],
                    s4 + gb2[4], s5 + gb2[5], s6 + gb2[6], s7 + gb2[7] };
    float mx = sv[0];
    #pragma unroll
    for (int j = 1; j < 8; ++j) mx = fmaxf(mx, sv[j]);
    float tot = 0.f, ex[8];
    #pragma unroll
    for (int j = 0; j < 8; ++j) { ex[j] = __expf(sv[j] - mx); tot += ex[j]; }
    float inv = 1.0f / tot;
    #pragma unroll
    for (int j = 0; j < 8; ++j) coeff[(size_t)row * 8 + j] = ex[j] * inv;
  }
}

// ---------------------------------------------------------------------------
// launch
// ---------------------------------------------------------------------------
extern "C" void kernel_launch(void* const* d_in, const int* in_sizes, int n_in,
                              void* d_out, int out_size, void* d_ws, size_t ws_size,
                              hipStream_t stream) {
  const float* z = (const float*)d_in[0];
  const float* c = (const float*)d_in[1];
  const float* w[5]   = {(const float*)d_in[2],  (const float*)d_in[6],
                         (const float*)d_in[10], (const float*)d_in[14],
                         (const float*)d_in[18]};
  const float* bb[5]  = {(const float*)d_in[3],  (const float*)d_in[7],
                         (const float*)d_in[11], (const float*)d_in[15],
                         (const float*)d_in[19]};
  const float* lng[5] = {(const float*)d_in[4],  (const float*)d_in[8],
                         (const float*)d_in[12], (const float*)d_in[16],
                         (const float*)d_in[20]};
  const float* lnb[5] = {(const float*)d_in[5],  (const float*)d_in[9],
                         (const float*)d_in[13], (const float*)d_in[17],
                         (const float*)d_in[21]};
  const float* gW0 = (const float*)d_in[22]; const float* gb0 = (const float*)d_in[23];
  const float* gW1 = (const float*)d_in[24]; const float* gb1 = (const float*)d_in[25];
  const float* gW2 = (const float*)d_in[26]; const float* gb2 = (const float*)d_in[27];

  char* ws = (char*)d_ws;
  __bf16* zc    = (__bf16*)(ws);               // 4096*1280 bf16
  __bf16* xb    = (__bf16*)(ws + 10485760);    // 4096*1792 bf16
  __bf16* g1    = (__bf16*)(ws + 25165824);    // 4096*512 bf16
  __bf16* g2    = (__bf16*)(ws + 29360128);    // 4096*512 bf16
  float*  coeff = (float*) (ws + 33554432);    // 4096*8 f32
  float*  lo    = (float*) (ws + 33685504);    // 4096*1024 f32 (layer_out)
  __bf16* wtb   = (__bf16*)(ws + 50462720);    // transposed bf16 weights, reused
                                               // (max 8*1792*1024*2 = 29,360,128 B)

  // prep: zc (gate input) + x0 = LN(zc) (stride 1280, stored in xb)
  prep_kernel<<<4096, 256, 0, stream>>>(z, c, lng[0], lnb[0], zc, xb);

  // gate MLP (transpose weights -> bf16 GEMM)
  transpose_bf16<<<dim3(40, 16, 1), 256, 0, stream>>>(gW0, wtb, 1280, 512);
  mixgemm<1, false, true, true ><<<dim3(64, 4), 256, 0, stream>>>(zc, wtb, gb0, nullptr, nullptr, nullptr, g1, 1280, 512);
  transpose_bf16<<<dim3(16, 16, 1), 256, 0, stream>>>(gW1, wtb, 512, 512);
  mixgemm<1, false, true, true ><<<dim3(64, 4), 256, 0, stream>>>(g1, wtb, gb1, nullptr, nullptr, nullptr, g2, 512, 512);
  gate_softmax<<<512, 256, 0, stream>>>(g2, gW2, gb2, coeff);

  // layer 0: no residual, leaky
  transpose_bf16<<<dim3(40, 32, 8), 256, 0, stream>>>(w[0], wtb, 1280, 1024);
  mixgemm<8, false, true, false><<<dim3(64, 8), 256, 0, stream>>>(xb, wtb, bb[0], coeff, nullptr, lo, nullptr, 1280, 1024);

  // layers 1..3: LN(concat(z, lo)) -> mix -> leaky(lo + mix), in place on lo
  for (int i = 1; i <= 3; ++i) {
    ln_concat_kernel<<<4096, 256, 0, stream>>>(z, lo, lng[i], lnb[i], xb);
    transpose_bf16<<<dim3(56, 32, 8), 256, 0, stream>>>(w[i], wtb, 1792, 1024);
    mixgemm<8, true, true, false><<<dim3(64, 8), 256, 0, stream>>>(xb, wtb, bb[i], coeff, lo, lo, nullptr, 1792, 1024);
  }

  // final: LN -> mix, no residual/activation, fp32 out
  ln_concat_kernel<<<4096, 256, 0, stream>>>(z, lo, lng[4], lnb[4], xb);
  transpose_bf16<<<dim3(56, 16, 8), 256, 0, stream>>>(w[4], wtb, 1792, 512);
  mixgemm<8, false, false, false><<<dim3(64, 4), 256, 0, stream>>>(xb, wtb, bb[4], coeff, nullptr, (float*)d_out, nullptr, 1792, 512);
}
